// GroundingNet_46299747451450
// MI455X (gfx1250) — matrix operand order repaired and checked by reading, not compiled, not run on hardware
//
#include <hip/hip_runtime.h>
#include <hip/hip_bf16.h>

typedef __attribute__((ext_vector_type(16))) _Float16 v16h;
typedef __attribute__((ext_vector_type(8)))  _Float16 v8h;
typedef __attribute__((ext_vector_type(8)))  float    v8f;
typedef __attribute__((ext_vector_type(4)))  unsigned int u32x4;
typedef __attribute__((ext_vector_type(8)))  int     i32x8;
typedef __attribute__((ext_vector_type(4)))  int     i32x4;

union AB16 { v16h v; v8h h[2]; };

// ---------------------------------------------------------------- prep kernels

__global__ __launch_bounds__(256) void k_f32_to_f16(const float* __restrict__ x,
                                                    _Float16* __restrict__ y, int n) {
  int i = blockIdx.x * 256 + threadIdx.x;
  if (i < n) y[i] = (_Float16)x[i];
}

// W (K x N, row major f32)  ->  Wp (Npad x Kpad, row major f16, transposed, zero padded)
__global__ __launch_bounds__(256) void k_pack_w(const float* __restrict__ W,
                                                _Float16* __restrict__ Wp,
                                                int K, int N, int Kpad, int Npad) {
  int i = blockIdx.x * 256 + threadIdx.x;
  if (i >= Npad * Kpad) return;
  int n = i / Kpad;
  int k = i - n * Kpad;
  float v = (n < N && k < K) ? W[(size_t)k * N + n] : 0.0f;
  Wp[i] = (_Float16)v;
}

// Fold BN into per-column scale/offset: y = s*(xW) + o
__global__ void k_make_so(const float* g, const float* bnb, const float* rm,
                          const float* rv, const float* bias,
                          float* s, float* o, int N, int Npad) {
  int n = blockIdx.x * blockDim.x + threadIdx.x;
  if (n >= Npad) return;
  if (n >= N) { s[n] = 0.0f; o[n] = 0.0f; return; }
  if (g) {
    float sc = g[n] * rsqrtf(rv[n] + 1e-5f);
    s[n] = sc;
    o[n] = sc * bias[n] + bnb[n] - sc * rm[n];
  } else {
    s[n] = 1.0f;
    o[n] = bias[n];
  }
}

// ---------------------------------------------------------------- TDM-staged GEMM
// Block = 8 waves = 128 rows x 64 cols. B panel (64 cols x 64 k) is DMA'd into
// LDS by the Tensor Data Mover (double buffered, TENSORcnt-tracked), with the
// D# padding feature inserting 16B per 128B row for bank-conflict-free reads.
// Epilogue: relu(acc*scale[n] + offs[n]) -> f16.
__global__ __launch_bounds__(256) void k_wgemm_tdm(
    const _Float16* __restrict__ A, const _Float16* __restrict__ Bp,
    const float* __restrict__ scale, const float* __restrict__ offs,
    _Float16* __restrict__ outH, int M, int K, int Npad) {
  __shared__ _Float16 Bsh[2][64][72];   // 72 = 64 + 8 f16 pad (row stride 144B)

  const int lane = threadIdx.x & 31;
  const int wid  = threadIdx.x >> 5;
  const int colGroups = Npad >> 6;
  const int rb = blockIdx.x / colGroups;
  const int cg = blockIdx.x - rb * colGroups;
  const int g  = lane >> 4;
  const int lr = lane & 15;

  const _Float16* Arow = A + (size_t)(rb * 128 + wid * 16 + lr) * K;

  // ---- Tensor DMA descriptor (uniform, per ISA cdna5 ch.8) ----
  const unsigned long long gbase =
      (unsigned long long)(uintptr_t)(Bp + (size_t)cg * 64 * K);
  const unsigned int ldsA[2] = {
      (unsigned int)(uintptr_t)&Bsh[0][0][0],
      (unsigned int)(uintptr_t)&Bsh[1][0][0]};

  // group1: data_size=2B(code 1), pad_enable, pad_interval=32 dwords (code 4),
  // pad_amount=4 dwords (code 3), tensor 64x64, tile 64x64, dim0_stride=K.
  const i32x8 g1 = {
      (int)((1u << 16) | (1u << 20) | (4u << 22) | (3u << 25)),
      (int)(64u << 16),          // tensor_dim0[15:0] = 64
      (int)(64u << 16),          // tensor_dim1[15:0] = 64
      (int)(64u << 16),          // tile_dim0 = 64
      64,                        // tile_dim1 = 64, tile_dim2 = 0
      K,                         // tensor_dim0_stride (data_size units)
      0, 0};
  const i32x4 gz4 = {0, 0, 0, 0};
  const i32x8 gz8 = {0, 0, 0, 0, 0, 0, 0, 0};

  auto tdm_issue = [&](int buf, int kc) {
    unsigned long long ga = gbase + (unsigned long long)kc * 2;  // bytes
    u32x4 g0;
    g0[0] = 1u;                                  // count = 1 valid descriptor
    g0[1] = ldsA[buf];                           // lds_addr
    g0[2] = (unsigned int)ga;                    // global_addr[31:0]
    g0[3] = (unsigned int)((ga >> 32) & 0x1FFFFFFu) | (2u << 30);  // [56:32] | type=2
    __builtin_amdgcn_tensor_load_to_lds(g0, g1, gz4, gz4, gz8, 0);
  };

  v8f zero = {};
  v8f acc[4];
#pragma unroll
  for (int t = 0; t < 4; ++t) acc[t] = zero;

  if (wid == 0) tdm_issue(0, 0);   // scalar-branched: TDM ignores EXEC

  const int nChunk = K >> 6;
  for (int i = 0; i < nChunk; ++i) {
    const int kc = i << 6;
    if (wid == 0) __builtin_amdgcn_s_wait_tensorcnt(0);
    __syncthreads();   // publish buf[i&1]; also fences last reads of buf[(i+1)&1]
    if (wid == 0 && i + 1 < nChunk) tdm_issue((i + 1) & 1, kc + 64);

    const _Float16* Bbuf = &Bsh[i & 1][0][0];
#pragma unroll
    for (int ks = 0; ks < 64; ks += 32) {
      AB16 a;
      a.h[0] = *(const v8h*)(Arow + kc + ks + g * 16);
      a.h[1] = *(const v8h*)(Arow + kc + ks + g * 16 + 8);
      if (ks == 32 && kc + 64 < K) __builtin_prefetch(Arow + kc + 64, 0, 3);
#pragma unroll
      for (int t = 0; t < 4; ++t) {
        AB16 bm;
        const _Float16* bl = Bbuf + (size_t)(t * 16 + lr) * 72 + ks + g * 16;
        bm.h[0] = *(const v8h*)bl;
        bm.h[1] = *(const v8h*)(bl + 8);
        acc[t] = __builtin_amdgcn_wmma_f32_16x16x32_f16(
            false, a.v, false, bm.v, (short)0, acc[t], false, false);
      }
    }
  }

#pragma unroll
  for (int t = 0; t < 4; ++t) {
    int n = cg * 64 + t * 16 + lr;
    float s = scale[n], o = offs[n];
#pragma unroll
    for (int r = 0; r < 8; ++r) {
      float v = fmaxf(acc[t][r] * s + o, 0.0f);
      int m = rb * 128 + wid * 16 + g * 8 + r;
      outH[(size_t)m * Npad + n] = (_Float16)v;
    }
  }
}

// ---------------------------------------------------------------- generic wave-tile GEMM
// (used for the tiny N=16 sigmoid head)
template <int NT, int ACT, bool OUT_F32>
__global__ __launch_bounds__(256) void k_wgemm(
    const _Float16* __restrict__ A, const _Float16* __restrict__ Bp,
    const float* __restrict__ scale, const float* __restrict__ offs,
    float* __restrict__ outF, _Float16* __restrict__ outH,
    int M, int K, int Npad, int outStride, int outCols) {
  const int lane = threadIdx.x & 31;
  const int wid  = (blockIdx.x * 256 + threadIdx.x) >> 5;
  const int colGroups = Npad / (NT * 16);
  const int rowTile = wid / colGroups;
  const int cg = wid - rowTile * colGroups;
  if (rowTile * 16 >= M) return;
  const int g  = lane >> 4;
  const int lr = lane & 15;

  const _Float16* Arow  = A  + (size_t)(rowTile * 16 + lr) * K;
  const _Float16* Bbase = Bp + (size_t)(cg * (NT * 16) + lr) * K + g * 16;

  v8f zero = {};
  v8f acc[NT];
#pragma unroll
  for (int t = 0; t < NT; ++t) acc[t] = zero;

  for (int k = 0; k < K; k += 32) {
    AB16 a;
    a.h[0] = *(const v8h*)(Arow + k + g * 16);
    a.h[1] = *(const v8h*)(Arow + k + g * 16 + 8);
    if (k + 32 < K) __builtin_prefetch(Arow + k + 32, 0, 3);
#pragma unroll
    for (int t = 0; t < NT; ++t) {
      AB16 b;
      const _Float16* bl = Bbase + (size_t)t * 16 * K + k;
      b.h[0] = *(const v8h*)(bl);
      b.h[1] = *(const v8h*)(bl + 8);
      acc[t] = __builtin_amdgcn_wmma_f32_16x16x32_f16(
          false, a.v, false, b.v, (short)0, acc[t], false, false);
    }
  }

#pragma unroll
  for (int t = 0; t < NT; ++t) {
    int n = cg * (NT * 16) + t * 16 + lr;
    float s = scale[n], o = offs[n];
#pragma unroll
    for (int r = 0; r < 8; ++r) {
      float v = acc[t][r] * s + o;
      if (ACT == 0) v = fmaxf(v, 0.0f);
      else if (ACT == 1) v = 1.0f / (1.0f + __expf(-v));
      int m = rowTile * 16 + g * 8 + r;
      if (n < outCols) {
        if (OUT_F32) outF[(size_t)m * outStride + n] = v;
        else         outH[(size_t)m * outStride + n] = (_Float16)v;
      }
    }
  }
}

// ---------------------------------------------------------------- fused edge MLP
__global__ __launch_bounds__(256) void k_edge_mlp(
    const float* __restrict__ bboxes, const float* __restrict__ dirs,
    const float* __restrict__ prios, const int* __restrict__ eidx,
    const _Float16* __restrict__ W1, const float* __restrict__ b1,
    const _Float16* __restrict__ W2, const float* __restrict__ b2,
    const _Float16* __restrict__ W3, const float* __restrict__ b3,
    float* __restrict__ out) {
  const int Nn = 512, E = 32768;
  __shared__ _Float16 attr[8][16][32];
  __shared__ _Float16 hbuf[8][16][72];

  const int wid  = threadIdx.x >> 5;
  const int lane = threadIdx.x & 31;
  const int gw   = blockIdx.x * 8 + wid;
  const int edge0 = gw * 16;
  const int b     = edge0 / E;
  const int ebase = edge0 - b * E;
  const int g = lane >> 4, lr = lane & 15;

  if (lane < 16) {
    int e = ebase + lane;
    const int* ep = eidx + (size_t)b * 2 * E;
    int src = ep[e];
    int dst = ep[E + e];
    const float inv = 1.0f / 1024.0f;
    _Float16* row = attr[wid][lane];
    const float* bb = bboxes + ((size_t)b * Nn + src) * 4;
    const float* dd = dirs   + ((size_t)b * Nn + src) * 4;
    row[0] = (_Float16)(bb[0] * inv); row[1] = (_Float16)(bb[1] * inv);
    row[2] = (_Float16)(bb[2] * inv); row[3] = (_Float16)(bb[3] * inv);
    row[4] = (_Float16)dd[0]; row[5] = (_Float16)dd[1];
    row[6] = (_Float16)dd[2]; row[7] = (_Float16)dd[3];
    row[8] = (_Float16)prios[(size_t)b * Nn + src];
    bb = bboxes + ((size_t)b * Nn + dst) * 4;
    dd = dirs   + ((size_t)b * Nn + dst) * 4;
    row[9]  = (_Float16)(bb[0] * inv); row[10] = (_Float16)(bb[1] * inv);
    row[11] = (_Float16)(bb[2] * inv); row[12] = (_Float16)(bb[3] * inv);
    row[13] = (_Float16)dd[0]; row[14] = (_Float16)dd[1];
    row[15] = (_Float16)dd[2]; row[16] = (_Float16)dd[3];
    row[17] = (_Float16)prios[(size_t)b * Nn + dst];
#pragma unroll
    for (int c = 18; c < 32; ++c) row[c] = (_Float16)0.0f;
  }
  // same-wave LDS ops are in-order (DScnt) -> no barrier needed

  v8f zero = {};
  AB16 a;

  // layer 1: (16x32) @ (32x64)
  a.h[0] = *(const v8h*)(&attr[wid][lr][g * 16]);
  a.h[1] = *(const v8h*)(&attr[wid][lr][g * 16 + 8]);
  v8f acc[4];
#pragma unroll
  for (int t = 0; t < 4; ++t) {
    AB16 bm;
    const _Float16* bl = W1 + (size_t)(t * 16 + lr) * 32 + g * 16;
    bm.h[0] = *(const v8h*)bl;
    bm.h[1] = *(const v8h*)(bl + 8);
    acc[t] = __builtin_amdgcn_wmma_f32_16x16x32_f16(
        false, a.v, false, bm.v, (short)0, zero, false, false);
  }
#pragma unroll
  for (int t = 0; t < 4; ++t) {
    int n = t * 16 + lr;
    float bv = b1[n];
#pragma unroll
    for (int r = 0; r < 8; ++r)
      hbuf[wid][g * 8 + r][n] = (_Float16)fmaxf(acc[t][r] + bv, 0.0f);
  }

  // layer 2: (16x64) @ (64x64)
  v8f acc2[4];
#pragma unroll
  for (int t = 0; t < 4; ++t) acc2[t] = zero;
#pragma unroll
  for (int kk = 0; kk < 64; kk += 32) {
    a.h[0] = *(const v8h*)(&hbuf[wid][lr][kk + g * 16]);
    a.h[1] = *(const v8h*)(&hbuf[wid][lr][kk + g * 16 + 8]);
#pragma unroll
    for (int t = 0; t < 4; ++t) {
      AB16 bm;
      const _Float16* bl = W2 + (size_t)(t * 16 + lr) * 64 + kk + g * 16;
      bm.h[0] = *(const v8h*)bl;
      bm.h[1] = *(const v8h*)(bl + 8);
      acc2[t] = __builtin_amdgcn_wmma_f32_16x16x32_f16(
          false, a.v, false, bm.v, (short)0, acc2[t], false, false);
    }
  }
#pragma unroll
  for (int t = 0; t < 4; ++t) {
    int n = t * 16 + lr;
    float bv = b2[n];
#pragma unroll
    for (int r = 0; r < 8; ++r)
      hbuf[wid][g * 8 + r][n] = (_Float16)fmaxf(acc2[t][r] + bv, 0.0f);
  }

  // layer 3: (16x64) @ (64x16), EC=10 live columns
  v8f acc3 = zero;
#pragma unroll
  for (int kk = 0; kk < 64; kk += 32) {
    a.h[0] = *(const v8h*)(&hbuf[wid][lr][kk + g * 16]);
    a.h[1] = *(const v8h*)(&hbuf[wid][lr][kk + g * 16 + 8]);
    AB16 bm;
    const _Float16* bl = W3 + (size_t)lr * 64 + kk + g * 16;
    bm.h[0] = *(const v8h*)bl;
    bm.h[1] = *(const v8h*)(bl + 8);
    acc3 = __builtin_amdgcn_wmma_f32_16x16x32_f16(
        false, a.v, false, bm.v, (short)0, acc3, false, false);
  }
  int n = lr;
  if (n < 10) {
    float bv = b3[n];
#pragma unroll
    for (int r = 0; r < 8; ++r) {
      float v = acc3[r] + bv;
      v = 1.0f / (1.0f + __expf(-v));
      int e = edge0 + g * 8 + r;
      out[(size_t)e * 10 + n] = v;
    }
  }
}

// ---------------------------------------------------------------- launcher

extern "C" void kernel_launch(void* const* d_in, const int* in_sizes, int n_in,
                              void* d_out, int out_size, void* d_ws, size_t ws_size,
                              hipStream_t stream) {
  (void)in_sizes; (void)n_in; (void)out_size; (void)ws_size;
  const int B = 32, Nn = 512, F = 1024, E = 32768;
  const int H1 = 512, H2 = 256, H3 = 128;
  const int M = B * Nn;  // 16384

  const float* roi   = (const float*)d_in[0];
  const float* bbox  = (const float*)d_in[1];
  const float* dirs  = (const float*)d_in[2];
  const float* prios = (const float*)d_in[3];
  const int*   eidx  = (const int*)d_in[4];
  const float* w1    = (const float*)d_in[5];
  const float* nb1   = (const float*)d_in[6];
  const float* bn1g  = (const float*)d_in[7];
  const float* bn1b  = (const float*)d_in[8];
  const float* bn1rm = (const float*)d_in[9];
  const float* bn1rv = (const float*)d_in[10];
  const float* w2    = (const float*)d_in[11];
  const float* nb2   = (const float*)d_in[12];
  const float* bn2g  = (const float*)d_in[13];
  const float* bn2b  = (const float*)d_in[14];
  const float* bn2rm = (const float*)d_in[15];
  const float* bn2rv = (const float*)d_in[16];
  const float* w3    = (const float*)d_in[17];
  const float* nb3   = (const float*)d_in[18];
  const float* niw   = (const float*)d_in[19];
  const float* nib   = (const float*)d_in[20];
  const float* epw1  = (const float*)d_in[21];
  const float* epb1  = (const float*)d_in[22];
  const float* epw2  = (const float*)d_in[23];
  const float* epb2  = (const float*)d_in[24];
  const float* eiw   = (const float*)d_in[25];
  const float* eib   = (const float*)d_in[26];

  char* ws = (char*)d_ws;
  size_t off = 0;
  auto alloc = [&](size_t bytes) -> void* {
    void* p = ws + off;
    off = (off + bytes + 255) & ~(size_t)255;
    return p;
  };
  _Float16* xh  = (_Float16*)alloc((size_t)M * F * 2);
  _Float16* h1  = (_Float16*)alloc((size_t)M * H1 * 2);
  _Float16* h2  = (_Float16*)alloc((size_t)M * H2 * 2);
  _Float16* h3  = (_Float16*)alloc((size_t)M * H3 * 2);
  _Float16* W1p = (_Float16*)alloc((size_t)H1 * F * 2);
  _Float16* W2p = (_Float16*)alloc((size_t)H2 * H1 * 2);
  _Float16* W3p = (_Float16*)alloc((size_t)H3 * H2 * 2);
  _Float16* Wnp = (_Float16*)alloc((size_t)16 * H3 * 2);
  _Float16* We1 = (_Float16*)alloc((size_t)64 * 32 * 2);
  _Float16* We2 = (_Float16*)alloc((size_t)64 * 64 * 2);
  _Float16* We3 = (_Float16*)alloc((size_t)16 * 64 * 2);
  float* s1 = (float*)alloc(512 * 4); float* o1 = (float*)alloc(512 * 4);
  float* s2 = (float*)alloc(256 * 4); float* o2 = (float*)alloc(256 * 4);
  float* s3 = (float*)alloc(128 * 4); float* o3 = (float*)alloc(128 * 4);
  float* sn = (float*)alloc(16 * 4);  float* on = (float*)alloc(16 * 4);

  // prep
  k_f32_to_f16<<<(M * F + 255) / 256, 256, 0, stream>>>(roi, xh, M * F);
  k_pack_w<<<(H1 * F + 255) / 256, 256, 0, stream>>>(w1, W1p, F, H1, F, H1);
  k_pack_w<<<(H2 * H1 + 255) / 256, 256, 0, stream>>>(w2, W2p, H1, H2, H1, H2);
  k_pack_w<<<(H3 * H2 + 255) / 256, 256, 0, stream>>>(w3, W3p, H2, H3, H2, H3);
  k_pack_w<<<(16 * H3 + 255) / 256, 256, 0, stream>>>(niw, Wnp, H3, 8, H3, 16);
  k_pack_w<<<(64 * 32 + 255) / 256, 256, 0, stream>>>(epw1, We1, 18, 64, 32, 64);
  k_pack_w<<<(64 * 64 + 255) / 256, 256, 0, stream>>>(epw2, We2, 64, 64, 64, 64);
  k_pack_w<<<(16 * 64 + 255) / 256, 256, 0, stream>>>(eiw, We3, 64, 10, 64, 16);
  k_make_so<<<2, 256, 0, stream>>>(bn1g, bn1b, bn1rm, bn1rv, nb1, s1, o1, 512, 512);
  k_make_so<<<1, 256, 0, stream>>>(bn2g, bn2b, bn2rm, bn2rv, nb2, s2, o2, 256, 256);
  k_make_so<<<1, 128, 0, stream>>>(nullptr, nullptr, nullptr, nullptr, nb3, s3, o3, 128, 128);
  k_make_so<<<1, 16, 0, stream>>>(nullptr, nullptr, nullptr, nullptr, nib, sn, on, 8, 16);

  // node MLP: TDM-staged GEMMs (128 rows x 64 cols per block)
  k_wgemm_tdm<<<(M / 128) * (H1 / 64), 256, 0, stream>>>(xh, W1p, s1, o1, h1, M, F, H1);
  k_wgemm_tdm<<<(M / 128) * (H2 / 64), 256, 0, stream>>>(h1, W2p, s2, o2, h2, M, H1, H2);
  k_wgemm_tdm<<<(M / 128) * (H3 / 64), 256, 0, stream>>>(h2, W3p, s3, o3, h3, M, H2, H3);
  k_wgemm<1, 1, true><<<128, 256, 0, stream>>>(h3, Wnp, sn, on, (float*)d_out, nullptr, M, H3, 16, 8, 8);

  // edge MLP: 1,048,576 edges, 128 per block
  float* edgeOut = (float*)d_out + (size_t)M * 8;
  k_edge_mlp<<<(B * E) / 128, 256, 0, stream>>>(bbox, dirs, prios, eidx,
                                                We1, epb1, We2, epb2, We3, eib, edgeOut);
}
